// DeformAlignNet_20925080666124
// MI455X (gfx1250) — matrix operand order repaired and checked by reading, not compiled, not run on hardware
//
#include <hip/hip_runtime.h>
#include <hip/hip_bf16.h>
#include <cstdint>

// ---------------- problem constants ----------------
#define BN_   8
#define CN_   128
#define HN_   64
#define WN_   64
#define HW_   (HN_*WN_)          // 4096
#define KK_   9
#define KDIM_ (CN_*KK_)          // 1152
#define MDIM_ (BN_*HW_)          // 32768
#define NDIM_ CN_                // 128
#define EPS_BN 1e-5f

typedef __attribute__((ext_vector_type(16))) __bf16 v16bf;
typedef __attribute__((ext_vector_type(8)))  __bf16 v8bf;
typedef __attribute__((ext_vector_type(8)))  float  v8f;

// =====================================================================
// 1) offset conv: out off[b][o][hw], o in 0..17, pad=1 3x3 over C=128.
//    one thread per (b,hw) computes all 18 channels: x taps loaded once,
//    weights are wave-uniform (scalar loads through K$).
// =====================================================================
__global__ void offconv_kernel(const float* __restrict__ x,
                               const float* __restrict__ w,    // [18][C][3][3]
                               const float* __restrict__ bias, // [18]
                               float* __restrict__ off)        // [B][18][HW]
{
    const int hw = blockIdx.x * 256 + threadIdx.x;
    const int b  = blockIdx.y;
    const int iy = hw >> 6, ix = hw & 63;

    float acc[18];
#pragma unroll
    for (int o = 0; o < 18; ++o) acc[o] = bias[o];

    const float* xb = x + (size_t)b * CN_ * HW_;
    for (int c = 0; c < CN_; ++c) {
        const float* xp = xb + (size_t)c * HW_;
        float t[9];
#pragma unroll
        for (int dy = 0; dy < 3; ++dy)
#pragma unroll
            for (int dx = 0; dx < 3; ++dx) {
                int yy = iy + dy - 1, xx = ix + dx - 1;
                bool v = (yy >= 0) & (yy < HN_) & (xx >= 0) & (xx < WN_);
                t[dy*3+dx] = v ? xp[yy*WN_ + xx] : 0.f;
            }
#pragma unroll
        for (int o = 0; o < 18; ++o) {
            const float* wo = w + ((size_t)o * CN_ + c) * 9;
#pragma unroll
            for (int k2 = 0; k2 < 9; ++k2)
                acc[o] = fmaf(t[k2], wo[k2], acc[o]);
        }
    }
    float* ob = off + (size_t)b * 18 * HW_;
#pragma unroll
    for (int o = 0; o < 18; ++o) ob[(size_t)o * HW_ + hw] = acc[o];
}

// =====================================================================
// 2) weight prep: Wb[(c*9+k)*128 + oc] = bf16( w[oc][c][k] )
// =====================================================================
__global__ void prep_wb_kernel(const float* __restrict__ w, __bf16* __restrict__ Wb)
{
    int idx = blockIdx.x * 256 + threadIdx.x;       // 0 .. 147455
    int kidx = idx >> 7;                            // c*9+k
    int oc   = idx & 127;
    int c = kidx / 9, k = kidx % 9;
    Wb[idx] = (__bf16)w[((size_t)oc * CN_ + c) * 9 + k];
}

// =====================================================================
// 3) bilinear sampling -> sampT[K= c*9+k][M= b*4096+hw]  (bf16, K-major
//    so stores are contiguous along hw)
// =====================================================================
__global__ void sample_kernel(const float* __restrict__ x,
                              const float* __restrict__ off,   // [B][18][HW]
                              __bf16* __restrict__ sampT)
{
    const int hw = blockIdx.x * 256 + threadIdx.x;
    const int c  = blockIdx.y;
    const int bz = blockIdx.z;          // b*9 + k
    const int b  = bz / 9, k = bz % 9;
    const int ky = k / 3, kx = k % 3;
    const int iy = hw >> 6, ix = hw & 63;

    const float dy = off[((size_t)b * 18 + 2*k    ) * HW_ + hw];
    const float dx = off[((size_t)b * 18 + 2*k + 1) * HW_ + hw];

    const float py = (float)(iy - 1 + ky) + dy;
    const float px = (float)(ix - 1 + kx) + dx;
    const float y0f = floorf(py), x0f = floorf(px);
    const float wy = py - y0f,    wx = px - x0f;
    const int y0 = (int)y0f, x0 = (int)x0f;

    const float* xp = x + ((size_t)b * CN_ + c) * HW_;
    auto g = [&](int yy, int xx) -> float {
        bool v = (yy >= 0) & (yy < HN_) & (xx >= 0) & (xx < WN_);
        int yc = min(max(yy, 0), HN_ - 1);
        int xc = min(max(xx, 0), WN_ - 1);
        float t = xp[yc * WN_ + xc];
        return v ? t : 0.f;
    };
    float s = g(y0,   x0  ) * (1.f-wy) * (1.f-wx)
            + g(y0,   x0+1) * (1.f-wy) * wx
            + g(y0+1, x0  ) * wy       * (1.f-wx)
            + g(y0+1, x0+1) * wy       * wx;

    sampT[(size_t)(c * 9 + k) * MDIM_ + (size_t)b * HW_ + hw] = (__bf16)s;
}

// =====================================================================
// 4) WMMA GEMM:  Y[b][oc][hw] = sum_K sampT[K][m] * Wb[K][oc] + bias[oc]
//    256 threads = 8 waves; tile 64M x 128N; K step 32.
//    Double-buffered LDS ping-pong: global loads for tile k+1 are issued
//    first, fenced with a sched_barrier so the scheduler cannot sink them
//    below the WMMA block, then tile k is computed from LDS while those
//    loads are in flight. One workgroup barrier per K step.
//    LDS holds A as [m][k] (stride 40 halves) and B as [n][k] (stride 40),
//    matching the ISA per-lane fragment layouts for 16-bit WMMA.
// =====================================================================
#define ABYTES_ 5120                     // 64*40*2
#define BUFB_   15360                    // (64*40 + 128*40)*2

__global__ void __launch_bounds__(256)
gemm_kernel(const __bf16* __restrict__ A,   // [KDIM][MDIM]
            const __bf16* __restrict__ Bw,  // [KDIM][128]
            const float*  __restrict__ bias,// [128]
            float* __restrict__ Y)          // [B][C][HW]
{
    __shared__ __align__(16) unsigned char smem[32768];

    const int tid  = threadIdx.x;
    const int lane = tid & 31;
    const int wv   = tid >> 5;
    const int m0   = blockIdx.x * 64;

    v8f acc0 = {}, acc1 = {}, acc2 = {}, acc3 = {};

    const int lk  = tid >> 3;            // 0..31 (K within tile)
    const int lma = (tid & 7) * 8;       // A m-offset (8 halves)
    const int lnb = (tid & 7) * 16;      // B n-offset (16 halves)

    // fragment indices (per the ISA 16-bit WMMA VGPR layout tables)
    const int row = lane & 15;
    const int akb = (lane < 16) ? 0 : 8;    // A: K0-7/K16-23 vs K8-15/K24-31
    const int ncol = wv * 16 + row;
    const int bkb = (lane < 16) ? 0 : 8;    // B: K0-7/K16-23 vs K8-15/K24-31

    v8bf ra, rb0, rb1;                   // global-load staging registers

    auto gload = [&](int k0) {
        ra  = *(const v8bf*)(A  + (size_t)(k0 + lk) * MDIM_ + m0 + lma);
        rb0 = *(const v8bf*)(Bw + (size_t)(k0 + lk) * 128 + lnb);
        rb1 = *(const v8bf*)(Bw + (size_t)(k0 + lk) * 128 + lnb + 8);
    };
    auto lstore = [&](int p) {
        __bf16* As = (__bf16*)(smem + p * BUFB_);
        __bf16* Bs = (__bf16*)(smem + p * BUFB_ + ABYTES_);
#pragma unroll
        for (int i = 0; i < 8; ++i) As[(lma + i) * 40 + lk] = ra[i];
#pragma unroll
        for (int i = 0; i < 8; ++i) Bs[(lnb + i) * 40 + lk]     = rb0[i];
#pragma unroll
        for (int i = 0; i < 8; ++i) Bs[(lnb + 8 + i) * 40 + lk] = rb1[i];
    };

    constexpr int NSTEP = KDIM_ / 32;    // 36
    gload(0);
    lstore(0);
    __syncthreads();

    for (int it = 0; it < NSTEP; ++it) {
        const int p = it & 1;
        if (it + 1 < NSTEP) gload((it + 1) * 32);   // in flight during WMMAs
#if __has_builtin(__builtin_amdgcn_sched_barrier)
        __builtin_amdgcn_sched_barrier(0);          // keep loads issued here
#endif

        const __bf16* As = (const __bf16*)(smem + p * BUFB_);
        const __bf16* Bs = (const __bf16*)(smem + p * BUFB_ + ABYTES_);

        // hoist all fragment loads so they pipeline under one dscnt wait
        union { v8bf h[2]; v16bf v; } bf, af0, af1, af2, af3;
        bf.h[0]  = *(const v8bf*)(Bs + ncol * 40 + bkb);
        bf.h[1]  = *(const v8bf*)(Bs + ncol * 40 + bkb + 16);
        af0.h[0] = *(const v8bf*)(As + (  0 + row) * 40 + akb);
        af0.h[1] = *(const v8bf*)(As + (  0 + row) * 40 + akb + 16);
        af1.h[0] = *(const v8bf*)(As + ( 16 + row) * 40 + akb);
        af1.h[1] = *(const v8bf*)(As + ( 16 + row) * 40 + akb + 16);
        af2.h[0] = *(const v8bf*)(As + ( 32 + row) * 40 + akb);
        af2.h[1] = *(const v8bf*)(As + ( 32 + row) * 40 + akb + 16);
        af3.h[0] = *(const v8bf*)(As + ( 48 + row) * 40 + akb);
        af3.h[1] = *(const v8bf*)(As + ( 48 + row) * 40 + akb + 16);

        acc0 = __builtin_amdgcn_wmma_f32_16x16x32_bf16(false, af0.v, false, bf.v, (short)0, acc0, false, false);
        acc1 = __builtin_amdgcn_wmma_f32_16x16x32_bf16(false, af1.v, false, bf.v, (short)0, acc1, false, false);
        acc2 = __builtin_amdgcn_wmma_f32_16x16x32_bf16(false, af2.v, false, bf.v, (short)0, acc2, false, false);
        acc3 = __builtin_amdgcn_wmma_f32_16x16x32_bf16(false, af3.v, false, bf.v, (short)0, acc3, false, false);

        if (it + 1 < NSTEP) lstore(1 - p);
        __syncthreads();
    }

    // ---- bias (each lane's column n is fixed across all 8 acc slots) ----
    const float bsv = bias[ncol];
#pragma unroll
    for (int r = 0; r < 8; ++r) { acc0[r] += bsv; acc1[r] += bsv; acc2[r] += bsv; acc3[r] += bsv; }

    // ---- stage through LDS as [n][m] so NCHW stores are coalesced ----
    float* Os = (float*)smem;               // [128][64]
    const int mb = (lane < 16) ? 0 : 8;
#pragma unroll
    for (int r = 0; r < 8; ++r) {
        Os[ncol * 64 +      mb + r] = acc0[r];
        Os[ncol * 64 + 16 + mb + r] = acc1[r];
        Os[ncol * 64 + 32 + mb + r] = acc2[r];
        Os[ncol * 64 + 48 + mb + r] = acc3[r];
    }
    __syncthreads();

    const int bi  = m0 >> 12;               // batch
    const int hw0 = m0 & 4095;
    const int nrow = tid >> 1;
    const int mo   = (tid & 1) * 32;
    float* yp = Y + ((size_t)bi * CN_ + nrow) * HW_ + hw0 + mo;
    const float* os = Os + nrow * 64 + mo;
#pragma unroll
    for (int i = 0; i < 32; i += 4)
        *(float4*)(yp + i) = *(const float4*)(os + i);
}

// =====================================================================
// 5) per-channel BN statistics: stats[c] = {mean, rsqrt(var+eps)}
// =====================================================================
__global__ void bn_stats_kernel(const float* __restrict__ Y, float* __restrict__ stats)
{
    const int c = blockIdx.x;
    const int tid = threadIdx.x, lane = tid & 31, wv = tid >> 5;
    float s = 0.f, s2 = 0.f;
    for (int i = tid; i < BN_ * HW_; i += 256) {
        int b = i >> 12, hw = i & 4095;
        float v = Y[((size_t)b * CN_ + c) * HW_ + hw];
        s += v; s2 += v * v;
    }
#pragma unroll
    for (int o = 16; o > 0; o >>= 1) {
        s  += __shfl_xor(s,  o, 32);
        s2 += __shfl_xor(s2, o, 32);
    }
    __shared__ float rs[8], rs2[8];
    if (lane == 0) { rs[wv] = s; rs2[wv] = s2; }
    __syncthreads();
    if (tid == 0) {
        float S = 0.f, S2 = 0.f;
#pragma unroll
        for (int i = 0; i < 8; ++i) { S += rs[i]; S2 += rs2[i]; }
        const float inv = 1.f / (float)(BN_ * HW_);
        float m  = S * inv;
        float vr = S2 * inv - m * m;
        stats[2*c]   = m;
        stats[2*c+1] = rsqrtf(vr + EPS_BN);
    }
}

// =====================================================================
// 6) BN apply (+optional residual) + ReLU
// =====================================================================
__global__ void bn_apply_kernel(const float* __restrict__ Y,
                                const float* __restrict__ stats,
                                const float* __restrict__ g,
                                const float* __restrict__ bb,
                                const float* __restrict__ res,  // may be null
                                float* __restrict__ out,
                                int with_res)
{
    const size_t i = (size_t)blockIdx.x * 256 + threadIdx.x;
    const int c = (int)((i >> 12) & 127);
    float m = stats[2*c], rstd = stats[2*c+1];
    float v = g[c] * (Y[i] - m) * rstd + bb[c];
    if (with_res) v += res[i];
    out[i] = fmaxf(v, 0.f);
}

// =====================================================================
// host orchestration
// =====================================================================
extern "C" void kernel_launch(void* const* d_in, const int* in_sizes, int n_in,
                              void* d_out, int out_size, void* d_ws, size_t ws_size,
                              hipStream_t stream)
{
    const float* x      = (const float*)d_in[0];
    const float* w_off  = (const float*)d_in[1];  // (L,2,18,C,3,3)
    const float* b_off  = (const float*)d_in[2];  // (L,2,18)
    const float* w_conv = (const float*)d_in[3];  // (L,2,C,C,3,3)
    const float* b_conv = (const float*)d_in[4];  // (L,2,C)
    const float* bn_g   = (const float*)d_in[5];  // (L,2,C)
    const float* bn_b   = (const float*)d_in[6];  // (L,2,C)
    float* out = (float*)d_out;

    unsigned char* ws = (unsigned char*)d_ws;
    float*  off   = (float*) (ws);                         //  2,359,296 B
    __bf16* Wb    = (__bf16*)(ws + 2359296);               //    294,912 B
    __bf16* sampT = (__bf16*)(ws + 2654208);               // 75,497,472 B
    float*  ybuf  = (float*) (ws + 78151680);              // 16,777,216 B
    float*  Tbuf  = (float*) (ws + 94928896);              // 16,777,216 B
    float*  Abuf  = (float*) (ws + 111706112);             // 16,777,216 B
    float*  stats = (float*) (ws + 128483328);             //      1,024 B

    auto stage = [&](const float* xin, int l, int s, const float* res, float* dst) {
        const int   ls  = l * 2 + s;
        const float* wo = w_off  + (size_t)ls * 18 * CN_ * 9;
        const float* bo = b_off  + (size_t)ls * 18;
        const float* wc = w_conv + (size_t)ls * CN_ * CN_ * 9;
        const float* bc = b_conv + (size_t)ls * CN_;
        const float* g  = bn_g   + (size_t)ls * CN_;
        const float* bb = bn_b   + (size_t)ls * CN_;

        offconv_kernel<<<dim3(HW_/256, BN_), 256, 0, stream>>>(xin, wo, bo, off);
        prep_wb_kernel<<<(KDIM_*NDIM_)/256, 256, 0, stream>>>(wc, Wb);
        sample_kernel<<<dim3(HW_/256, CN_, BN_*KK_), 256, 0, stream>>>(xin, off, sampT);
        gemm_kernel<<<MDIM_/64, 256, 0, stream>>>(sampT, Wb, bc, ybuf);
        bn_stats_kernel<<<CN_, 256, 0, stream>>>(ybuf, stats);
        bn_apply_kernel<<<(BN_*CN_*HW_)/256, 256, 0, stream>>>(
            ybuf, stats, g, bb, res, dst, res != nullptr ? 1 : 0);
    };

    // l = 0
    stage(x,    0, 0, nullptr, Tbuf);   // T = relu(bn(dc(x)))
    stage(Tbuf, 0, 1, x,       Abuf);   // A = relu(bn(dc(T)) + x)
    // l = 1
    stage(Abuf, 1, 0, nullptr, Tbuf);   // T = relu(bn(dc(A)))
    stage(Tbuf, 1, 1, Abuf,    out);    // out = relu(bn(dc(T)) + A)
}
